// RIPGeo_21801253994576
// MI455X (gfx1250) — matrix operand order repaired and checked by reading, not compiled, not run on hardware
//
#include <hip/hip_runtime.h>
#include <hip/hip_bf16.h>

typedef __attribute__((ext_vector_type(16))) _Float16 v16h;
typedef __attribute__((ext_vector_type(8)))  _Float16 v8h;
typedef __attribute__((ext_vector_type(8)))  float    v8f;

#define K_NN 5
#define COS_EPS 1e-8f
#define KNN_T 128

static __device__ __forceinline__ v16h cat8h(v8h lo, v8h hi) {
    return __builtin_shufflevector(lo, hi,
        0, 1, 2, 3, 4, 5, 6, 7, 8, 9, 10, 11, 12, 13, 14, 15);
}

// ---------------------------------------------------------------------------
// e[n][j] = b_emb[j] + sum_d feats[n][d] * W_emb[j][d]
// blockDim.x == D+1 == E == 64; one block per node row.
// ---------------------------------------------------------------------------
__global__ __launch_bounds__(64) void emb_kernel(
    const float* __restrict__ lm_X, const float* __restrict__ tg_X,
    const float* __restrict__ lm_delay, const float* __restrict__ tg_delay,
    const float* __restrict__ W_emb, const float* __restrict__ b_emb,
    float* __restrict__ e, int N1, int D)
{
    __shared__ float feat[64];
    int n = blockIdx.x;
    int j = threadIdx.x;
    if (n < N1) {
        if (j < D)  feat[j] = lm_X[(size_t)n * D + j];
        if (j == D) feat[j] = lm_delay[n];
    } else {
        int t = n - N1;
        if (j < D)  feat[j] = tg_X[(size_t)t * D + j];
        if (j == D) feat[j] = tg_delay[t];
    }
    __syncthreads();
    float s = b_emb[j];
    const float* wrow = W_emb + (size_t)j * (D + 1);
#pragma unroll 8
    for (int d = 0; d <= D; ++d) s = fmaf(feat[d], wrow[d], s);
    e[(size_t)n * 64 + j] = s;
}

// ---------------------------------------------------------------------------
// Per (n,h): U[h][n][d] = f16(e[n][d]*w1[d][h]), V likewise with w2;
// na[h][n] = ||u||, nb[h][n] = ||v||.  64 threads, one per d.
// ---------------------------------------------------------------------------
__global__ __launch_bounds__(64) void head_pack_kernel(
    const float* __restrict__ e, const float* __restrict__ w1,
    const float* __restrict__ w2,
    _Float16* __restrict__ U, _Float16* __restrict__ V,
    float* __restrict__ na, float* __restrict__ nb, int N, int H)
{
    int n = blockIdx.x, h = blockIdx.y, d = threadIdx.x;
    float ev = e[(size_t)n * 64 + d];
    float u = ev * w1[d * H + h];
    float v = ev * w2[d * H + h];
    size_t base = ((size_t)h * N + n) * 64 + d;
    U[base] = (_Float16)u;
    V[base] = (_Float16)v;
    __shared__ float su[64], sv[64];
    su[d] = u * u;
    sv[d] = v * v;
    __syncthreads();
    for (int s = 32; s > 0; s >>= 1) {
        if (d < s) { su[d] += su[d + s]; sv[d] += sv[d + s]; }
        __syncthreads();
    }
    if (d == 0) {
        na[(size_t)h * N + n] = sqrtf(su[0]);
        nb[(size_t)h * N + n] = sqrtf(sv[0]);
    }
}

// ---------------------------------------------------------------------------
// 5-NN over lm_X rows. One block per row i; each thread keeps a local top-5
// over its strided slice of j, thread 0 merges in LDS.
// ---------------------------------------------------------------------------
__global__ __launch_bounds__(KNN_T) void knn_kernel(
    const float* __restrict__ lm_X, int* __restrict__ nbr, int N1, int D)
{
    __shared__ float xi[64];
    __shared__ float cd[KNN_T * K_NN];
    __shared__ int   ci[KNN_T * K_NN];
    int i = blockIdx.x;
    int t = threadIdx.x;
    for (int d = t; d < D; d += KNN_T) xi[d] = lm_X[(size_t)i * D + d];
    __syncthreads();

    float bd[K_NN];
    int   bi[K_NN];
#pragma unroll
    for (int q = 0; q < K_NN; ++q) { bd[q] = __builtin_huge_valf(); bi[q] = -1; }

    for (int j = t; j < N1; j += KNN_T) {
        if (j == i) continue;
        const float* xj = lm_X + (size_t)j * D;
        float s = 0.0f;
#pragma unroll 7
        for (int d = 0; d < D; ++d) { float df = xi[d] - xj[d]; s = fmaf(df, df, s); }
        if (s < bd[K_NN - 1]) {
            int p = K_NN - 1;
            while (p > 0 && bd[p - 1] > s) { bd[p] = bd[p - 1]; bi[p] = bi[p - 1]; --p; }
            bd[p] = s; bi[p] = j;
        }
    }
#pragma unroll
    for (int q = 0; q < K_NN; ++q) { cd[t * K_NN + q] = bd[q]; ci[t * K_NN + q] = bi[q]; }
    __syncthreads();
    if (t == 0) {
        for (int q = 0; q < K_NN; ++q) {
            float best = __builtin_huge_valf(); int bj = -1, bp = -1;
            for (int p = 0; p < KNN_T * K_NN; ++p)
                if (cd[p] < best) { best = cd[p]; bj = ci[p]; bp = p; }
            if (bp >= 0) cd[bp] = __builtin_huge_valf();
            nbr[i * K_NN + q] = bj;
        }
    }
}

// ---------------------------------------------------------------------------
// y_pred[i] = mean of lm_Y over 5 neighbours
// ---------------------------------------------------------------------------
__global__ void ypred_kernel(const float* __restrict__ lm_Y,
                             const int* __restrict__ nbr,
                             float* __restrict__ out, int N1)
{
    int i = blockIdx.x * blockDim.x + threadIdx.x;
    if (i >= N1) return;
    float s0 = 0.0f, s1 = 0.0f;
#pragma unroll
    for (int q = 0; q < K_NN; ++q) {
        int j = nbr[i * K_NN + q];
        s0 += lm_Y[j * 2];
        s1 += lm_Y[j * 2 + 1];
    }
    out[i * 2]     = s0 * (1.0f / K_NN);
    out[i * 2 + 1] = s1 * (1.0f / K_NN);
}

// ---------------------------------------------------------------------------
// adj_teacher: lm/lm block = knn adjacency, tg/tg block = 0, cross = 1.
// float4 stores (N1 and N divisible by 4).
// ---------------------------------------------------------------------------
__global__ void teacher_kernel(const int* __restrict__ nbr,
                               float* __restrict__ T, int N, int N1)
{
    int ncols4 = N >> 2;
    int idx = blockIdx.x * blockDim.x + threadIdx.x;
    if (idx >= N * ncols4) return;
    int i  = idx / ncols4;
    int j0 = (idx - i * ncols4) << 2;
    float4 v;
    float* vp = &v.x;
    if (i < N1) {
        if (j0 < N1) {
            int b0 = nbr[i * K_NN + 0], b1 = nbr[i * K_NN + 1], b2 = nbr[i * K_NN + 2];
            int b3 = nbr[i * K_NN + 3], b4 = nbr[i * K_NN + 4];
#pragma unroll
            for (int c = 0; c < 4; ++c) {
                int j = j0 + c;
                vp[c] = (j == b0 || j == b1 || j == b2 || j == b3 || j == b4) ? 1.0f : 0.0f;
            }
        } else {
            v = make_float4(1.0f, 1.0f, 1.0f, 1.0f);
        }
    } else {
        float x = (j0 < N1) ? 1.0f : 0.0f;   // tg row: 1 toward lm cols, 0 in tg block
        v = make_float4(x, x, x, x);
    }
    *(float4*)(T + (size_t)i * N + j0) = v;
}

// ---------------------------------------------------------------------------
// adj[n,m] = mean_h sigmoid( dot_h(n,m) / max(na*nb, eps) )
// One wave per 32x32 output tile (2x2 WMMA tiles): A/B operands reused twice,
// 8 chained v_wmma_f32_16x16x32_f16 per head (K=64), 64 WMMAs when H=8 is
// fully unrolled.  Epilogue uses v_rcp_f32 instead of IEEE division.
//
// A (16-bit 16x32): lane L holds row M=L&15; K chunks [kb,kb+8) and
// [kb+16,kb+24), kb = 8*(L>=16)  -> two contiguous v8h loads.
// B (16-bit 32x16): lane L holds col N=L&15; K chunks [kb,kb+8) and [kb+16,kb+24) as A,
// kb = 8*(L>=16) -> two contiguous v8h loads.
// C/D: lane L covers col N=L&15; VGPR v covers row M = v + 8*(L>=16).
// ---------------------------------------------------------------------------
template <int HC>
__global__ __launch_bounds__(32) void adj_wmma_kernel(
    const _Float16* __restrict__ U, const _Float16* __restrict__ V,
    const float* __restrict__ na, const float* __restrict__ nb,
    float* __restrict__ adj, int N, int Hdyn)
{
    const int H    = (HC > 0) ? HC : Hdyn;
    const int tm   = blockIdx.x * 32;
    const int tn   = blockIdx.y * 32;
    const int lane = threadIdx.x;
    const int hi16 = lane >> 4;          // 0 or 1
    const int l15  = lane & 15;
    const int kbA  = hi16 * 8;
    const int kbB  = hi16 * 8;

    v8f acc[2][2] = {};
#pragma unroll
    for (int h = 0; h < H; ++h) {
        // ---- load A (2 m-tiles x 2 k-steps) ----
        v16h a[2][2];
#pragma unroll
        for (int mi = 0; mi < 2; ++mi) {
            const _Float16* Uh = U + ((size_t)h * N + tm + mi * 16 + l15) * 64;
            v8h lo0 = *(const v8h*)(Uh + kbA);
            v8h hi0 = *(const v8h*)(Uh + kbA + 16);
            v8h lo1 = *(const v8h*)(Uh + kbA + 32);
            v8h hi1 = *(const v8h*)(Uh + kbA + 48);
            a[mi][0] = cat8h(lo0, hi0);
            a[mi][1] = cat8h(lo1, hi1);
        }
        // ---- load B (2 n-tiles x 2 k-steps) ----
        v16h b[2][2];
#pragma unroll
        for (int ni = 0; ni < 2; ++ni) {
            const _Float16* Vh = V + ((size_t)h * N + tn + ni * 16 + l15) * 64;
            b[ni][0] = cat8h(*(const v8h*)(Vh + kbB), *(const v8h*)(Vh + kbB + 16));        // K = [kbB,kbB+8) , [kbB+16,kbB+24)
            b[ni][1] = cat8h(*(const v8h*)(Vh + kbB + 32), *(const v8h*)(Vh + kbB + 48));   // K = 32 + same
        }
        // ---- 2x2 tiles: WMMA + fused sigmoid epilogue ----
#pragma unroll
        for (int mi = 0; mi < 2; ++mi) {
            v8f nav = *(const v8f*)(na + (size_t)h * N + tm + mi * 16 + hi16 * 8);
#pragma unroll
            for (int ni = 0; ni < 2; ++ni) {
                float nbv = nb[(size_t)h * N + tn + ni * 16 + l15];
                v8f c = {};
                c = __builtin_amdgcn_wmma_f32_16x16x32_f16(false, a[mi][0], false,
                        b[ni][0], (short)0, c, false, false);
                c = __builtin_amdgcn_wmma_f32_16x16x32_f16(false, a[mi][1], false,
                        b[ni][1], (short)0, c, false, false);
#pragma unroll
                for (int v = 0; v < 8; ++v) {
                    float denom = fmaxf(nav[v] * nbv, COS_EPS);
                    float sim   = c[v] * __builtin_amdgcn_rcpf(denom);
                    float ex    = __expf(-sim);
                    acc[mi][ni][v] += __builtin_amdgcn_rcpf(1.0f + ex);
                }
            }
        }
    }
    const float inv_h = 1.0f / (float)H;
#pragma unroll
    for (int mi = 0; mi < 2; ++mi)
#pragma unroll
        for (int ni = 0; ni < 2; ++ni)
#pragma unroll
            for (int v = 0; v < 8; ++v) {
                int row = tm + mi * 16 + hi16 * 8 + v;
                int col = tn + ni * 16 + l15;
                adj[(size_t)row * N + col] = acc[mi][ni][v] * inv_h;
            }
}

// ---------------------------------------------------------------------------
extern "C" void kernel_launch(void* const* d_in, const int* in_sizes, int n_in,
                              void* d_out, int out_size, void* d_ws, size_t ws_size,
                              hipStream_t stream) {
    const float* lm_X     = (const float*)d_in[0];
    const float* lm_Y     = (const float*)d_in[1];
    const float* tg_X     = (const float*)d_in[2];
    // const float* tg_Y  = (const float*)d_in[3];   // unused by outputs
    const float* lm_delay = (const float*)d_in[4];
    const float* tg_delay = (const float*)d_in[5];
    const float* W_emb    = (const float*)d_in[6];
    const float* b_emb    = (const float*)d_in[7];
    const float* w1       = (const float*)d_in[8];
    const float* w2       = (const float*)d_in[9];

    const int N1 = in_sizes[1] / 2;          // 2000
    const int N2 = in_sizes[3] / 2;          // 2000
    const int D  = in_sizes[0] / N1;         // 63
    const int E  = in_sizes[6] / (D + 1);    // 64  (kernels assume 64)
    const int H  = in_sizes[8] / E;          // 8
    const int N  = N1 + N2;                  // 4000

    // ---- workspace layout (256B aligned chunks) ----
    char* ws = (char*)d_ws;
    size_t p = 0;
    auto take = [&](size_t bytes) -> char* {
        char* r = ws + p;
        p += (bytes + 255) & ~(size_t)255;
        return r;
    };
    float*    e_buf = (float*)   take((size_t)N * 64 * sizeof(float));
    _Float16* U_buf = (_Float16*)take((size_t)H * N * 64 * sizeof(_Float16));
    _Float16* V_buf = (_Float16*)take((size_t)H * N * 64 * sizeof(_Float16));
    float*    na    = (float*)   take((size_t)H * N * sizeof(float));
    float*    nb    = (float*)   take((size_t)H * N * sizeof(float));
    int*      nbr   = (int*)     take((size_t)N1 * K_NN * sizeof(int));

    // ---- output layout: y_pred [N1*2] | adj [N*N] | adj_teacher [N*N] ----
    float* out_y       = (float*)d_out;
    float* out_adj     = out_y + (size_t)N1 * 2;
    float* out_teacher = out_adj + (size_t)N * N;

    // 1) embeddings
    emb_kernel<<<N, E, 0, stream>>>(lm_X, tg_X, lm_delay, tg_delay,
                                    W_emb, b_emb, e_buf, N1, D);
    // 2) per-head operand packing + norms
    head_pack_kernel<<<dim3(N, H), 64, 0, stream>>>(e_buf, w1, w2,
                                                    U_buf, V_buf, na, nb, N, H);
    // 3) 5-NN over landmarks
    knn_kernel<<<N1, KNN_T, 0, stream>>>(lm_X, nbr, N1, D);
    // 4) y_pred
    ypred_kernel<<<(N1 + 255) / 256, 256, 0, stream>>>(lm_Y, nbr, out_y, N1);
    // 5) teacher adjacency (64 MB, float4 stores)
    {
        int total = N * (N >> 2);
        teacher_kernel<<<(total + 255) / 256, 256, 0, stream>>>(nbr, out_teacher, N, N1);
    }
    // 6) fused multi-head cosine-sigmoid adjacency via WMMA (32x32 per wave)
    dim3 grid(N / 32, N / 32);
    if (H == 8) {
        adj_wmma_kernel<8><<<grid, 32, 0, stream>>>(U_buf, V_buf, na, nb,
                                                    out_adj, N, H);
    } else {
        adj_wmma_kernel<0><<<grid, 32, 0, stream>>>(U_buf, V_buf, na, nb,
                                                    out_adj, N, H);
    }
}
